// RandomReductionLinear_34952443855185
// MI455X (gfx1250) — compile-verified, matrix operand order checked by reading, not run on hardware
//
#include <hip/hip_runtime.h>
#include <cstdint>
#include <cstddef>

#define B_DIM 2048
#define I_DIM 4096
#define O_DIM 2048
#define S_DIM 64

typedef unsigned int   u32;
typedef unsigned short u16;
typedef __attribute__((ext_vector_type(4)))  u32   u32x4;
typedef __attribute__((ext_vector_type(8)))  u32   u32x8;
typedef __attribute__((ext_vector_type(8)))  int   i32x8;
typedef __attribute__((ext_vector_type(4)))  int   i32x4;
typedef __attribute__((ext_vector_type(16))) __bf16 v16bf;
typedef __attribute__((ext_vector_type(8)))  float v8f;

__device__ __forceinline__ u16 f32_to_bf16(float f) {
    u32 u = __builtin_bit_cast(u32, f);
    u += 0x7FFFu + ((u >> 16) & 1u);   // round-to-nearest-even
    return (u16)(u >> 16);
}
__device__ __forceinline__ float bf16_to_f32(u16 h) {
    u32 u = ((u32)h) << 16;
    return __builtin_bit_cast(float, u);
}

// ---------------------------------------------------------------------------
// Phase 1: zero the densified-transposed weight matrix WdT[O][I] (bf16)
// ---------------------------------------------------------------------------
__global__ __launch_bounds__(256) void rrl_zero(uint4* __restrict__ p) {
    size_t i = (size_t)blockIdx.x * 256 + threadIdx.x;
    uint4 z; z.x = 0u; z.y = 0u; z.z = 0u; z.w = 0u;
    p[i] = z;
}

// ---------------------------------------------------------------------------
// Phase 2: scatter (perm, weight) into WdT. One thread owns one output row o,
// so duplicate indices within a row accumulate correctly with no atomics.
// ---------------------------------------------------------------------------
__global__ __launch_bounds__(256) void rrl_scatter(const int* __restrict__ perm,
                                                   const float* __restrict__ weight,
                                                   u16* __restrict__ wdT) {
    int o = blockIdx.x * 256 + threadIdx.x;
    if (o >= O_DIM) return;
    u16* row = wdT + (size_t)o * I_DIM;
    const int*   p = perm   + (size_t)o * S_DIM;
    const float* w = weight + (size_t)o * S_DIM;
    for (int s = 0; s < S_DIM; ++s) {
        int idx = p[s];
        row[idx] = f32_to_bf16(bf16_to_f32(row[idx]) + w[s]);
    }
}

// ---------------------------------------------------------------------------
// Phase 3: Y[B][O] = X[B][I](->bf16) * WdT[O][I]^T + bias, via WMMA bf16.
// Workgroup tile: 128(M) x 128(N), BK=64. 8 waves: waveM in {0,1} (64 rows),
// waveN in {0..3} (32 cols); each wave holds 4x2 f32 16x16 accumulators.
// B tile is DMA'd into LDS by the Tensor Data Mover (wave 0), A tile staged
// by all threads with f32->bf16 conversion.
// ---------------------------------------------------------------------------
__global__ __launch_bounds__(256) void rrl_gemm(const float* __restrict__ x,
                                                const u16*   __restrict__ wdT,
                                                const float* __restrict__ bias,
                                                float* __restrict__ out) {
    __shared__ u16 tileA[128 * 64];
    __shared__ u16 tileB[128 * 64];

    const int tid   = threadIdx.x;
    const int wave  = tid >> 5;
    const int lane  = tid & 31;
    const int lm    = lane & 15;       // M (A/D) or N (B/D) index within tile
    const int half  = lane >> 4;       // selects k-groups per ISA 16-bit layout
    const int waveM = wave >> 2;       // 0..1  -> 64-row slab
    const int waveN = wave & 3;        // 0..3  -> 32-col slab
    const int mBase = blockIdx.y * 128;
    const int nBase = blockIdx.x * 128;

    const int ccol = tid & 15;         // float4 column within BK for A staging
    const int rrow = tid >> 4;         // 0..15 base row for A staging

    v8f acc[4][2] = {};

    for (int k0 = 0; k0 < I_DIM; k0 += 64) {
        // ---- wave 0: issue TDM load of B tile (128 n-rows x 64 k, bf16) ----
        if (wave == 0) {
            unsigned long long ga =
                (unsigned long long)(uintptr_t)(wdT + (size_t)nBase * I_DIM + k0);
            u32 ldsOff = (u32)(uintptr_t)(void*)tileB;   // low 32 bits = LDS offset
            u32x4 g0;
            g0[0] = 1u;                                   // count = 1 valid descriptor
            g0[1] = ldsOff;                               // lds_addr
            g0[2] = (u32)ga;                              // global_addr[31:0]
            g0[3] = (u32)(ga >> 32) | 0x80000000u;        // addr[56:32] | type=2
            i32x8 g1;
            g1[0] = 0x00010000;                           // data_size = 2 bytes
            g1[1] = (int)(((u32)I_DIM & 0xFFFFu) << 16);  // tensor_dim0 lo16
            g1[2] = (int)(((u32)I_DIM >> 16) | (((u32)O_DIM & 0xFFFFu) << 16));
            g1[3] = (int)(((u32)O_DIM >> 16) | (64u << 16));  // tile_dim0 = 64
            g1[4] = 128;                                  // tile_dim1 = 128 rows
            g1[5] = I_DIM;                                // tensor_dim0_stride
            g1[6] = 0;
            g1[7] = 0;
            i32x4 z4 = {0, 0, 0, 0};
            i32x8 z8 = {0, 0, 0, 0, 0, 0, 0, 0};
            __builtin_amdgcn_tensor_load_to_lds(g0, g1, z4, z4, z8, 0);
        }

        // ---- all waves: stage A tile (128 m x 64 k), f32 -> bf16 ----
        #pragma unroll
        for (int j = 0; j < 8; ++j) {
            int row = rrow + j * 16;
            const float4 v = *(const float4*)(x + (size_t)(mBase + row) * I_DIM
                                                + k0 + ccol * 4);
            uint2 pv;
            pv.x = (u32)f32_to_bf16(v.x) | ((u32)f32_to_bf16(v.y) << 16);
            pv.y = (u32)f32_to_bf16(v.z) | ((u32)f32_to_bf16(v.w) << 16);
            *(uint2*)&tileA[row * 64 + ccol * 4] = pv;
        }

        if (wave == 0) __builtin_amdgcn_s_wait_tensorcnt(0);
        __syncthreads();

        // ---- compute: two k-chunks of 32 ----
        #pragma unroll
        for (int kk = 0; kk < 2; ++kk) {
            v16bf afrag[4];
            v16bf bfrag[2];
            #pragma unroll
            for (int mt = 0; mt < 4; ++mt) {
                int e = (waveM * 64 + mt * 16 + lm) * 64 + kk * 32 + half * 8;
                uint4 lo = *(const uint4*)&tileA[e];
                uint4 hi = *(const uint4*)&tileA[e + 16];
                u32x8 t = {lo.x, lo.y, lo.z, lo.w, hi.x, hi.y, hi.z, hi.w};
                afrag[mt] = __builtin_bit_cast(v16bf, t);
            }
            #pragma unroll
            for (int nt = 0; nt < 2; ++nt) {
                int e = (waveN * 32 + nt * 16 + lm) * 64 + kk * 32 + half * 8;
                uint4 lo = *(const uint4*)&tileB[e];
                uint4 hi = *(const uint4*)&tileB[e + 16];
                u32x8 t = {lo.x, lo.y, lo.z, lo.w, hi.x, hi.y, hi.z, hi.w};
                bfrag[nt] = __builtin_bit_cast(v16bf, t);
            }
            #pragma unroll
            for (int mt = 0; mt < 4; ++mt) {
                #pragma unroll
                for (int nt = 0; nt < 2; ++nt) {
                    acc[mt][nt] = __builtin_amdgcn_wmma_f32_16x16x32_bf16(
                        false, afrag[mt], false, bfrag[nt],
                        (short)0, acc[mt][nt], false, false);
                }
            }
        }
        __syncthreads();
    }

    // ---- epilogue: D layout (lane: n=lane%16; vgpr r: m = r + 8*(lane/16)) ----
    #pragma unroll
    for (int nt = 0; nt < 2; ++nt) {
        int col = nBase + waveN * 32 + nt * 16 + lm;
        float bv = bias[col];
        #pragma unroll
        for (int mt = 0; mt < 4; ++mt) {
            #pragma unroll
            for (int r = 0; r < 8; ++r) {
                int row = mBase + waveM * 64 + mt * 16 + r + 8 * half;
                out[(size_t)row * O_DIM + col] = acc[mt][nt][r] + bv;
            }
        }
    }
}

// ---------------------------------------------------------------------------
// Fallback (only if d_ws is too small): LDS-staged gather + scalar FMAs.
// ---------------------------------------------------------------------------
__global__ __launch_bounds__(256) void rrl_gather(const float* __restrict__ x,
                                                  const int* __restrict__ perm,
                                                  const float* __restrict__ weight,
                                                  const float* __restrict__ bias,
                                                  float* __restrict__ out) {
    __shared__ float xrow[I_DIM];
    int b = blockIdx.x;
    for (int i = threadIdx.x; i < I_DIM; i += 256)
        xrow[i] = x[(size_t)b * I_DIM + i];
    __syncthreads();
    for (int o = threadIdx.x; o < O_DIM; o += 256) {
        float acc = bias[o];
        const int*   p = perm   + (size_t)o * S_DIM;
        const float* w = weight + (size_t)o * S_DIM;
        #pragma unroll 8
        for (int s = 0; s < S_DIM; ++s)
            acc += xrow[p[s]] * w[s];
        out[(size_t)b * O_DIM + o] = acc;
    }
}

// ---------------------------------------------------------------------------
extern "C" void kernel_launch(void* const* d_in, const int* in_sizes, int n_in,
                              void* d_out, int out_size, void* d_ws, size_t ws_size,
                              hipStream_t stream) {
    const float* x      = (const float*)d_in[0];
    const int*   perm   = (const int*)  d_in[1];
    const float* weight = (const float*)d_in[2];
    const float* bias   = (const float*)d_in[3];
    float*       out    = (float*)d_out;

    const size_t need = (size_t)O_DIM * I_DIM * sizeof(u16);  // 16 MB for WdT
    if (ws_size >= need) {
        u16* wdT = (u16*)d_ws;
        const int n16 = (int)(need / 16);                     // uint4 count
        rrl_zero<<<n16 / 256, 256, 0, stream>>>((uint4*)d_ws);
        rrl_scatter<<<O_DIM / 256, 256, 0, stream>>>(perm, weight, wdT);
        rrl_gemm<<<dim3(O_DIM / 128, B_DIM / 128), 256, 0, stream>>>(x, wdT, bias, out);
    } else {
        rrl_gather<<<B_DIM, 256, 0, stream>>>(x, perm, weight, bias, out);
    }
}
